// CharCNNEncoder_39694087749662
// MI455X (gfx1250) — matrix operand order repaired and checked by reading, not compiled, not run on hardware
//
#include <hip/hip_runtime.h>
#include <hip/hip_bf16.h>

typedef __attribute__((ext_vector_type(16))) _Float16 v16h;
typedef __attribute__((ext_vector_type(8)))  _Float16 v8h;
typedef __attribute__((ext_vector_type(8)))  float    v8f;

#define N_TABLES 9          // (k=2: p=0,1) (k=3: p=0..2) (k=4: p=0..3)
#define TBL_C    128        // vocab
#define TBL_F    64         // filters padded 50 -> 64
#define TBL_ELEMS (N_TABLES * TBL_C * TBL_F)   // 73728 floats = 288 KB LDS
#define WPB      64         // words per block
#define NTHREADS 256        // 8 waves

// d_ws layout (bytes):
//   [0, 8192)            Apad  : f16 [128 chars][32 K]       (emb, K padded 30->32)
//   [8192, 45056)        Bpad  : f16 [9 t][64 f][32 K]       (weights, padded+transposed)
//   [45056, 47360)       BiasP : f32 [9 t][64 f]             (bias on p==0 tables, else 0)
#define APAD_HALFS  (128 * 32)
#define BPAD_HALFS  (N_TABLES * 64 * 32)
#define BPAD_BYTE_OFF 8192
#define BIAS_BYTE_OFF 45056
#define BIAS_FLOATS  (N_TABLES * 64)

__device__ __forceinline__ void decode_table(int t, int& ksz, int& p,
                                             const float*& wk, const float*& bk,
                                             const float* w2, const float* b2,
                                             const float* w3, const float* b3,
                                             const float* w4, const float* b4) {
    if (t < 2)      { ksz = 2; p = t;     wk = w2; bk = b2; }
    else if (t < 5) { ksz = 3; p = t - 2; wk = w3; bk = b3; }
    else            { ksz = 4; p = t - 5; wk = w4; bk = b4; }
}

// Tiny one-block kernel: zero-pad + transpose WMMA operands into d_ws so the
// hot kernel's operand loads are unconditional, contiguous, aligned vector loads.
__global__ __launch_bounds__(NTHREADS)
void charcnn_pack_kernel(const float* __restrict__ emb,
                         const float* __restrict__ w2, const float* __restrict__ b2,
                         const float* __restrict__ w3, const float* __restrict__ b3,
                         const float* __restrict__ w4, const float* __restrict__ b4,
                         _Float16* __restrict__ apad, _Float16* __restrict__ bpad,
                         float* __restrict__ biasp)
{
    const int tid = threadIdx.x;
    // Apad[c][d]
    for (int i = tid; i < APAD_HALFS; i += NTHREADS) {
        const int c = i >> 5, d = i & 31;
        apad[i] = (_Float16)(d < 30 ? emb[c * 30 + d] : 0.0f);
    }
    // Bpad[t][f][d] = w_k[f, d, p]
    for (int i = tid; i < BPAD_HALFS; i += NTHREADS) {
        const int d = i & 31;
        const int f = (i >> 5) & 63;
        const int t = i >> 11;
        int ksz, p; const float* wk; const float* bk;
        decode_table(t, ksz, p, wk, bk, w2, b2, w3, b3, w4, b4);
        float v = 0.0f;
        if (d < 30 && f < 50) v = wk[(f * 30 + d) * ksz + p];
        bpad[i] = (_Float16)v;
    }
    // BiasP[t][f]
    for (int i = tid; i < BIAS_FLOATS; i += NTHREADS) {
        const int f = i & 63;
        const int t = i >> 6;
        int ksz, p; const float* wk; const float* bk;
        decode_table(t, ksz, p, wk, bk, w2, b2, w3, b3, w4, b4);
        biasp[i] = (p == 0 && f < 50) ? bk[f] : 0.0f;
    }
}

// Main kernel. Phase 1: branch-free WMMA table build into LDS.
// Phase 2: conv+relu+maxpool as LDS gather-adds.
__global__ __launch_bounds__(NTHREADS)
void charcnn_tbl_kernel(const int* __restrict__ x,
                        const _Float16* __restrict__ apad,
                        const _Float16* __restrict__ bpad,
                        const float* __restrict__ biasp,
                        float* __restrict__ out, int nWords)
{
    extern __shared__ float tbl[];   // [9][128][64]

    const int tid   = threadIdx.x;
    const int lane  = tid & 31;
    const int wid_s = __builtin_amdgcn_readfirstlane(tid >> 5);   // 0..7, SGPR

    // ---------------- Phase 1: WMMA table build ----------------
    // tile = t*32 + mt*4 + nt ; 8 waves x 36 iters = 288 tiles.
    const int m  = lane & 15;
    const int ah = (lane < 16) ? 0 : 16;    // A K-base (contiguous K-halves)
    const int bh = (lane < 16) ? 0 : 16;    // B K-base
    const int mh = lane >> 4;               // C row half

    for (int i = 0; i < 36; ++i) {
        const int tile = wid_s + 8 * i;     // SGPR
        const int t  = tile >> 5;
        const int mt = (tile >> 2) & 7;
        const int nt = tile & 3;
        const int f  = nt * 16 + (lane & 15);

        // A row: two contiguous 16B chunks (K {ah..ah+7} and {ah+8..ah+15})
        const _Float16* arow = apad + (mt * 16 + m) * 32;
        const v8h a_lo = *(const v8h*)(arow + ah);
        const v8h a_hi = *(const v8h*)(arow + ah + 8);
        v16h a;
        #pragma unroll
        for (int q = 0; q < 8; ++q) { a[q] = a_lo[q]; a[8 + q] = a_hi[q]; }

        // B column f: one contiguous 32B chunk (K bh..bh+15)
        const v16h bmat = *(const v16h*)(bpad + ((t * 64 + f) * 32) + bh);

        v8f c = {};
        c = __builtin_amdgcn_wmma_f32_16x16x32_f16(
                /*neg_a=*/false, a, /*neg_b=*/false, bmat,
                /*c_mod=*/(short)0, c, /*reuse_a=*/false, /*reuse_b=*/false);

        const float bias = biasp[t * 64 + f];
        #pragma unroll
        for (int r = 0; r < 8; ++r) {
            const int crow = mt * 16 + r + 8 * mh;
            tbl[(t * TBL_C + crow) * TBL_F + f] = c[r] + bias;
        }
    }
    __syncthreads();

    // ---------------- Phase 2: gather-add conv/maxpool ----------------
    const int pair_s = wid_s >> 1;           // word slot 0..3 (SGPR)
    const int hi_s   = wid_s & 1;            // filter half (SGPR)
    const int f2     = hi_s * 32 + lane;     // 0..63; only f2<50 stored

    const int base = blockIdx.x * WPB;
    for (int it = 0; it < WPB / 4; ++it) {
        const int word = base + it * 4 + pair_s;   // SGPR
        if (word >= nWords) continue;

        __builtin_prefetch(x + (size_t)(word + 4) * 24, 0, 1);

        int ids[24];
        const int4* xq = (const int4*)(x + (size_t)word * 24);  // 96B rows: 16B aligned
        #pragma unroll
        for (int u = 0; u < 6; ++u) {
            const int4 q = xq[u];
            ids[4*u]   = q.x; ids[4*u+1] = q.y;
            ids[4*u+2] = q.z; ids[4*u+3] = q.w;
        }
        int off[24];
        #pragma unroll
        for (int j = 0; j < 24; ++j) off[j] = ids[j] * TBL_F + f2;

        const int T = TBL_C * TBL_F;   // 8192 floats per table

        float m2 = -3.0e38f;
        #pragma unroll
        for (int j = 0; j < 23; ++j) {
            const float s = tbl[off[j]] + tbl[T + off[j + 1]];
            m2 = fmaxf(m2, s);
        }
        float m3 = -3.0e38f;
        #pragma unroll
        for (int j = 0; j < 22; ++j) {
            const float s = tbl[2*T + off[j]] + tbl[3*T + off[j+1]] + tbl[4*T + off[j+2]];
            m3 = fmaxf(m3, s);
        }
        float m4 = -3.0e38f;
        #pragma unroll
        for (int j = 0; j < 21; ++j) {
            const float s = (tbl[5*T + off[j]]   + tbl[6*T + off[j+1]])
                          + (tbl[7*T + off[j+2]] + tbl[8*T + off[j+3]]);
            m4 = fmaxf(m4, s);
        }

        // relu then max == clamp pooled max at 0
        m2 = fmaxf(m2, 0.0f); m3 = fmaxf(m3, 0.0f); m4 = fmaxf(m4, 0.0f);

        if (f2 < 50) {
            float* o = out + (size_t)word * 150;
            o[f2]       = m2;
            o[50 + f2]  = m3;
            o[100 + f2] = m4;
        }
    }
}

extern "C" void kernel_launch(void* const* d_in, const int* in_sizes, int n_in,
                              void* d_out, int out_size, void* d_ws, size_t ws_size,
                              hipStream_t stream) {
    (void)n_in; (void)ws_size; (void)out_size;
    const int*   x   = (const int*)  d_in[0];
    const float* emb = (const float*)d_in[1];
    const float* w2  = (const float*)d_in[2];
    const float* b2  = (const float*)d_in[3];
    const float* w3  = (const float*)d_in[4];
    const float* b3  = (const float*)d_in[5];
    const float* w4  = (const float*)d_in[6];
    const float* b4  = (const float*)d_in[7];
    float* out = (float*)d_out;

    char* ws = (char*)d_ws;
    _Float16* apad  = (_Float16*)ws;
    _Float16* bpad  = (_Float16*)(ws + BPAD_BYTE_OFF);
    float*    biasp = (float*)   (ws + BIAS_BYTE_OFF);

    const int nWords = in_sizes[0] / 24;                  // 32768
    const int grid   = (nWords + WPB - 1) / WPB;          // 512 blocks
    const size_t lds = (size_t)TBL_ELEMS * sizeof(float); // 294912 B

    charcnn_pack_kernel<<<1, NTHREADS, 0, stream>>>(
        emb, w2, b2, w3, b3, w4, b4, apad, bpad, biasp);
    charcnn_tbl_kernel<<<grid, NTHREADS, lds, stream>>>(
        x, apad, bpad, biasp, out, nWords);
}
